// Feature_extract_83940840833731
// MI455X (gfx1250) — matrix operand order repaired and checked by reading, not compiled, not run on hardware
//
#include <hip/hip_runtime.h>
#include <hip/hip_bf16.h>

typedef __attribute__((ext_vector_type(16))) _Float16 v16h;
typedef __attribute__((ext_vector_type(8)))  _Float16 v8h;
typedef __attribute__((ext_vector_type(8)))  float    v8f;

// ---------------------------------------------------------------------------
// Shared host/device parameter structs
// ---------------------------------------------------------------------------
struct LayerP {
  const _Float16* W;     // f16 weights, [coutPad16 rows][kPad32 cols], zero padded
  const float* bias;
  const float* gain;
  const float* beta;
  int cout;              // true output width
  int coutPad;           // ceil32(cout)  (K-dim padding for the NEXT layer)
};

struct BArgs {
  const float* feat;     // per-point features (p0 / p1cat / p2cat)
  const float* pts;      // per-point coords   (x0 / x1 / x2)
  const float* cent;     // centroids (new_xyz) -- unused in MODE 2
  const int*   gi;       // group indices (B,S,K) -- unused in MODE 2
  float*       out;      // pooled output
  int outStride, outOffset;
  int S, K, Npts, Cfeat, cin, cinPad;
  LayerP L0, L1, L2;
  int bufBOff, redOff;   // byte offsets inside dynamic LDS
};

// ---------------------------------------------------------------------------
// Small prep kernels
// ---------------------------------------------------------------------------
__global__ void transpose_b3n(const float* __restrict__ in, float* __restrict__ out,
                              int B, int N) {
  int t = blockIdx.x * blockDim.x + threadIdx.x;
  int total = B * 3 * N;
  if (t >= total) return;
  int n = t % N;
  int c = (t / N) % 3;
  int b = t / (3 * N);
  out[((size_t)b * N + n) * 3 + c] = in[t];
}

__global__ void wconv_kernel(const float* __restrict__ w, _Float16* __restrict__ dst,
                             int rows, int cols, int colsPad, int total) {
  int t = blockIdx.x * blockDim.x + threadIdx.x;
  if (t >= total) return;
  int r = t / colsPad;
  int c = t - r * colsPad;
  float v = (r < rows && c < cols) ? w[(size_t)r * cols + c] : 0.0f;
  dst[t] = (_Float16)v;
}

// ---------------------------------------------------------------------------
// Farthest point sampling: one block per batch. Emits new_xyz directly.
// Matches the scan semantics: output index s is the "far" BEFORE the update.
// ---------------------------------------------------------------------------
__global__ void __launch_bounds__(256) fps_kernel(const float* __restrict__ xyz,
                                                  float* __restrict__ newXyz,
                                                  int N, int np) {
  __shared__ float sv[256];
  __shared__ int   si[256];
  __shared__ int   curFar;
  const int b = blockIdx.x, tid = threadIdx.x;
  const float* X = xyz + (size_t)b * N * 3;
  if (tid == 0) curFar = 0;
  float dist[8];
#pragma unroll
  for (int i = 0; i < 8; ++i) dist[i] = 1e10f;
  __syncthreads();
  for (int s = 0; s < np; ++s) {
    int far = curFar;
    if (tid < 3) newXyz[((size_t)b * np + s) * 3 + tid] = X[(size_t)far * 3 + tid];
    float cx = X[(size_t)far * 3 + 0];
    float cy = X[(size_t)far * 3 + 1];
    float cz = X[(size_t)far * 3 + 2];
    float best = -1.0f;
    int   bidx = 0;
#pragma unroll
    for (int i = 0; i < 8; ++i) {
      int n = tid + (i << 8);
      if (n < N) {
        float dx = X[(size_t)n * 3 + 0] - cx;
        float dy = X[(size_t)n * 3 + 1] - cy;
        float dz = X[(size_t)n * 3 + 2] - cz;
        float d = dx * dx + dy * dy + dz * dz;
        float nd = fminf(dist[i], d);
        dist[i] = nd;
        if (nd > best || (nd == best && n < bidx)) { best = nd; bidx = n; }
      }
    }
    sv[tid] = best; si[tid] = bidx;
    __syncthreads();
    for (int off = 128; off > 0; off >>= 1) {
      if (tid < off) {
        float ov = sv[tid + off]; int oi = si[tid + off];
        if (ov > sv[tid] || (ov == sv[tid] && oi < si[tid])) { sv[tid] = ov; si[tid] = oi; }
      }
      __syncthreads();
    }
    if (tid == 0) curFar = si[0];
    __syncthreads();
  }
}

// ---------------------------------------------------------------------------
// Ball query: ascending-index scan == sort/truncate/pad-with-first semantics.
// One thread per centroid.
// ---------------------------------------------------------------------------
__global__ void ballq_kernel(const float* __restrict__ xyz, const float* __restrict__ cent,
                             int* __restrict__ gi, int B, int N, int S, int K, float r2) {
  int t = blockIdx.x * blockDim.x + threadIdx.x;
  if (t >= B * S) return;
  int b = t / S;
  const float* X = xyz + (size_t)b * N * 3;
  float cx = cent[(size_t)t * 3 + 0];
  float cy = cent[(size_t)t * 3 + 1];
  float cz = cent[(size_t)t * 3 + 2];
  int* g = gi + (size_t)t * K;
  int cnt = 0;
  for (int n = 0; n < N && cnt < K; ++n) {
    float dx = X[(size_t)n * 3 + 0] - cx;
    float dy = X[(size_t)n * 3 + 1] - cy;
    float dz = X[(size_t)n * 3 + 2] - cz;
    if (dx * dx + dy * dy + dz * dz <= r2) g[cnt++] = n;
  }
  if (cnt == 0) { g[0] = 0; cnt = 1; }  // centroid itself always qualifies; safety
  int f = g[0];
  for (int j = cnt; j < K; ++j) g[j] = f;
}

// ---------------------------------------------------------------------------
// WMMA tile loads (wave32, 16x16x32 f16). Layouts per CDNA5 ISA 7.12.2.
// A (16x32 f16): lane<16 row m=lane, K runs {base..base+7, base+16..base+23},
//                base = 8*(lane>>4). Two contiguous 16B LDS loads per lane.
// B (32x16 f16): lane holds column n=lane&15, K same as A (base..base+7, base+16..base+23).
//                Two contiguous 16B global loads per lane.
// ---------------------------------------------------------------------------
__device__ __forceinline__ v16h load_a_tile(const _Float16* __restrict__ X, int xs,
                                            int rowBase, int kb, int lane) {
  const _Float16* p = X + (size_t)(rowBase + (lane & 15)) * xs + kb + ((lane >> 4) << 3);
  v8h lo = *(const v8h*)(p);
  v8h hi = *(const v8h*)(p + 16);
  v16h a;
#pragma unroll
  for (int i = 0; i < 8; ++i) { a[i] = lo[i]; a[i + 8] = hi[i]; }
  return a;
}

__device__ __forceinline__ v16h load_b_tile(const _Float16* __restrict__ W, int ws,
                                            int colBase, int kb, int lane) {
  const _Float16* p = W + (size_t)(colBase + (lane & 15)) * ws + kb + ((lane >> 4) << 3);
  v8h lo = *(const v8h*)(p);
  v8h hi = *(const v8h*)(p + 16);
  v16h b;
#pragma unroll
  for (int i = 0; i < 8; ++i) { b[i] = lo[i]; b[i + 8] = hi[i]; }
  return b;
}

// One MLP layer: Y[rows][cout] = relu(g*(X*W^T + b) + beta), f16 out into LDS.
// Tiles are wave-uniform (EXEC all ones around every WMMA).
__device__ void mlp_layer(const _Float16* __restrict__ X, int xs, int kdim, int rows,
                          const LayerP& L, _Float16* __restrict__ Y, int ys) {
  const int lane = threadIdx.x & 31, wave = threadIdx.x >> 5, nw = blockDim.x >> 5;
  const int rtn = rows >> 4;
  const int ctn = (L.cout + 15) >> 4;
  for (int t = wave; t < rtn * ctn; t += nw) {
    int rt = t / ctn, ct = t - rt * ctn;
    v8f acc = {};
    for (int kb = 0; kb < kdim; kb += 32) {
      v16h av = load_a_tile(X, xs, rt << 4, kb, lane);
      v16h bv = load_b_tile(L.W, kdim, ct << 4, kb, lane);
      acc = __builtin_amdgcn_wmma_f32_16x16x32_f16(false, av, false, bv,
                                                   (short)0, acc, false, false);
    }
    int o = (ct << 4) + (lane & 15);
    float bi = 0.0f, gv = 0.0f, be = 0.0f;
    if (o < L.cout) { bi = L.bias[o]; gv = L.gain[o]; be = L.beta[o]; }
    int rb = (rt << 4) + ((lane >> 4) << 3);
#pragma unroll
    for (int r = 0; r < 8; ++r) {
      float v = gv * (acc[r] + bi) + be;
      v = v > 0.0f ? v : 0.0f;
      Y[(size_t)(rb + r) * ys + o] = (_Float16)v;
    }
  }
  // zero the K-padding columns [ctn*16, coutPad) so the next layer reads zeros
  const int padFrom = ctn << 4;
  if (L.coutPad > padFrom) {
    for (int r = wave; r < rows; r += nw)
      for (int c = padFrom + lane; c < L.coutPad; c += 32)
        Y[(size_t)r * ys + c] = (_Float16)0.0f;
  }
}

// Final layer fused with max-pool over rows. ReLU output >= 0, so float max
// == int max on the raw bits; red[] pre-initialized to 0 (== relu floor).
__device__ void mlp_layer_final(const _Float16* __restrict__ X, int xs, int kdim, int rows,
                                const LayerP& L, int* __restrict__ red) {
  const int lane = threadIdx.x & 31, wave = threadIdx.x >> 5, nw = blockDim.x >> 5;
  const int rtn = rows >> 4;
  const int ctn = L.cout >> 4;  // all final widths are multiples of 16
  for (int t = wave; t < rtn * ctn; t += nw) {
    int rt = t / ctn, ct = t - rt * ctn;
    v8f acc = {};
    for (int kb = 0; kb < kdim; kb += 32) {
      v16h av = load_a_tile(X, xs, rt << 4, kb, lane);
      v16h bv = load_b_tile(L.W, kdim, ct << 4, kb, lane);
      acc = __builtin_amdgcn_wmma_f32_16x16x32_f16(false, av, false, bv,
                                                   (short)0, acc, false, false);
    }
    int o = (ct << 4) + (lane & 15);
    float bi = L.bias[o], gv = L.gain[o], be = L.beta[o];
    float m = 0.0f;
#pragma unroll
    for (int r = 0; r < 8; ++r) {
      float v = gv * (acc[r] + bi) + be;
      v = v > 0.0f ? v : 0.0f;
      m = v > m ? v : m;
    }
    atomicMax(&red[o], __float_as_int(m));
  }
}

// ---------------------------------------------------------------------------
// One workgroup per centroid: gather group -> LDS (f16), 3 WMMA layers,
// fused max-pool, write pooled feature vector.
// MODE 0: SA1 (feat=p0[3], pts=x0, rel-xyz appended)
// MODE 1: SA2 (feat=p1cat[320], pts=x1, rel-xyz appended)
// MODE 2: SA3 (rows = all 128 centroids; feat order: xyz first, then p2[512])
// ---------------------------------------------------------------------------
template <int MODE>
__global__ void __launch_bounds__(256) branch_mlp(BArgs A) {
  extern __shared__ __align__(16) char smem[];
  _Float16* bufA = (_Float16*)smem;
  _Float16* bufB = (_Float16*)(smem + A.bufBOff);
  int*      red  = (int*)(smem + A.redOff);

  const int bs = blockIdx.x;
  const int b  = bs / A.S;
  const int cinLds = A.cinPad + 8;
  const int lane = threadIdx.x & 31, wave = threadIdx.x >> 5, nw = blockDim.x >> 5;

  float cx = 0.0f, cy = 0.0f, cz = 0.0f;
  if (MODE != 2) {
    const float* c = A.cent + (size_t)bs * 3;
    cx = c[0]; cy = c[1]; cz = c[2];
  }
  const int* g = (MODE == 2) ? (const int*)nullptr : (A.gi + (size_t)bs * A.K);

  // Gather group into bufA (f16), zero-fill channels [cin, cinPad)
  for (int row = wave; row < A.K; row += nw) {
    int src;
    if (MODE == 2) src = b * A.Npts + row;
    else           src = b * A.Npts + g[row];
    for (int c = lane; c < A.cinPad; c += 32) {
      float v = 0.0f;
      if (MODE == 2) {
        if (c < 3)            v = A.pts[(size_t)src * 3 + c];
        else if (c < A.cin)   v = A.feat[(size_t)src * A.Cfeat + (c - 3)];
      } else {
        if (c < A.Cfeat)      v = A.feat[(size_t)src * A.Cfeat + c];
        else if (c < A.cin) {
          int cc = c - A.Cfeat;
          float cv = (cc == 0) ? cx : ((cc == 1) ? cy : cz);
          v = A.pts[(size_t)src * 3 + cc] - cv;
        }
      }
      bufA[(size_t)row * cinLds + c] = (_Float16)v;
    }
  }
  for (int t = threadIdx.x; t < A.L2.cout; t += blockDim.x) red[t] = 0;
  __syncthreads();

  mlp_layer(bufA, cinLds, A.cinPad, A.K, A.L0, bufB, A.L0.coutPad + 8);
  __syncthreads();
  mlp_layer(bufB, A.L0.coutPad + 8, A.L0.coutPad, A.K, A.L1, bufA, A.L1.coutPad + 8);
  __syncthreads();
  mlp_layer_final(bufA, A.L1.coutPad + 8, A.L1.coutPad, A.K, A.L2, red);
  __syncthreads();

  float* outp = A.out + (size_t)bs * A.outStride + A.outOffset;
  for (int t = threadIdx.x; t < A.L2.cout; t += blockDim.x)
    outp[t] = __int_as_float(red[t]);
}

// ---------------------------------------------------------------------------
// Host orchestration
// ---------------------------------------------------------------------------
struct WDesc { int di, rows, cols; };
// d_in: 0=xyz 1=norm 2=cls_label, then 18 layers x (W,b,g,beta) in pytree order
static const WDesc g_wd[18] = {
  { 3, 32, 6 }, { 7, 32, 32 }, { 11, 64, 32 },        // sa1 branch0
  { 15, 64, 6 }, { 19, 64, 64 }, { 23, 128, 64 },     // sa1 branch1
  { 27, 64, 6 }, { 31, 96, 64 }, { 35, 128, 96 },     // sa1 branch2
  { 39, 128, 323 }, { 43, 128, 128 }, { 47, 256, 128 },// sa2 branch0
  { 51, 128, 323 }, { 55, 196, 128 }, { 59, 256, 196 },// sa2 branch1
  { 63, 256, 515 }, { 67, 512, 256 }, { 71, 1024, 512 }// sa3
};

static inline int ceil16(int x) { return (x + 15) & ~15; }
static inline int ceil32(int x) { return (x + 31) & ~31; }

extern "C" void kernel_launch(void* const* d_in, const int* in_sizes, int n_in,
                              void* d_out, int out_size, void* d_ws, size_t ws_size,
                              hipStream_t stream) {
  (void)in_sizes; (void)n_in; (void)out_size; (void)ws_size;
  const int B = 16, N = 2048;
  const float* xyz_raw = (const float*)d_in[0];
  const float* nrm_raw = (const float*)d_in[1];

  char* ws = (char*)d_ws;
  size_t off = 0;
  auto alloc = [&](size_t bytes) -> char* {
    char* p = ws + off;
    off = (off + bytes + 255) & ~(size_t)255;
    return p;
  };
  float* x0   = (float*)alloc((size_t)B * N * 3 * 4);
  float* p0   = (float*)alloc((size_t)B * N * 3 * 4);
  float* x1   = (float*)alloc((size_t)B * 512 * 3 * 4);
  float* x2   = (float*)alloc((size_t)B * 128 * 3 * 4);
  int*   gi1a = (int*)alloc((size_t)B * 512 * 32 * 4);
  int*   gi1b = (int*)alloc((size_t)B * 512 * 64 * 4);
  int*   gi1c = (int*)alloc((size_t)B * 512 * 128 * 4);
  int*   gi2a = (int*)alloc((size_t)B * 128 * 64 * 4);
  int*   gi2b = (int*)alloc((size_t)B * 128 * 128 * 4);
  float* p1c  = (float*)alloc((size_t)B * 512 * 320 * 4);
  float* p2c  = (float*)alloc((size_t)B * 128 * 512 * 4);

  // f16 weight staging
  size_t woff[18], wtot = 0;
  for (int i = 0; i < 18; ++i) {
    int rp = ceil16(g_wd[i].rows), cp = ceil32(g_wd[i].cols);
    woff[i] = wtot;
    wtot += (size_t)rp * cp;
  }
  _Float16* wbase = (_Float16*)alloc(wtot * 2);
  for (int i = 0; i < 18; ++i) {
    int rp = ceil16(g_wd[i].rows), cp = ceil32(g_wd[i].cols);
    int total = rp * cp;
    wconv_kernel<<<(total + 255) / 256, 256, 0, stream>>>(
        (const float*)d_in[g_wd[i].di], wbase + woff[i],
        g_wd[i].rows, g_wd[i].cols, cp, total);
  }

  // transpose (B,3,N) -> (B,N,3)
  {
    int total = B * 3 * N;
    transpose_b3n<<<(total + 255) / 256, 256, 0, stream>>>(xyz_raw, x0, B, N);
    transpose_b3n<<<(total + 255) / 256, 256, 0, stream>>>(nrm_raw, p0, B, N);
  }

  auto mkL = [&](int li) -> LayerP {
    LayerP L;
    L.W    = wbase + woff[li];
    L.bias = (const float*)d_in[g_wd[li].di + 1];
    L.gain = (const float*)d_in[g_wd[li].di + 2];
    L.beta = (const float*)d_in[g_wd[li].di + 3];
    L.cout = g_wd[li].rows;
    L.coutPad = ceil32(L.cout);
    return L;
  };

  auto runBranch = [&](int mode, int l0, const float* feat, const float* pts,
                       const float* cent, const int* gi, float* out,
                       int outStride, int outOffset, int S, int K, int Npts,
                       int Cfeat, int cin) {
    BArgs A;
    A.feat = feat; A.pts = pts; A.cent = cent; A.gi = gi; A.out = out;
    A.outStride = outStride; A.outOffset = outOffset;
    A.S = S; A.K = K; A.Npts = Npts; A.Cfeat = Cfeat;
    A.cin = cin; A.cinPad = ceil32(cin);
    A.L0 = mkL(l0); A.L1 = mkL(l0 + 1); A.L2 = mkL(l0 + 2);
    int s0 = A.cinPad > A.L1.coutPad ? A.cinPad : A.L1.coutPad;
    A.bufBOff = K * (s0 + 8) * 2;
    A.redOff  = A.bufBOff + K * (A.L0.coutPad + 8) * 2;
    int smem  = A.redOff + A.L2.cout * 4;
    int nb = B * S;
    if (mode == 0) {
      (void)hipFuncSetAttribute(reinterpret_cast<const void*>(&branch_mlp<0>),
                                hipFuncAttributeMaxDynamicSharedMemorySize, smem);
      branch_mlp<0><<<nb, 256, smem, stream>>>(A);
    } else if (mode == 1) {
      (void)hipFuncSetAttribute(reinterpret_cast<const void*>(&branch_mlp<1>),
                                hipFuncAttributeMaxDynamicSharedMemorySize, smem);
      branch_mlp<1><<<nb, 256, smem, stream>>>(A);
    } else {
      (void)hipFuncSetAttribute(reinterpret_cast<const void*>(&branch_mlp<2>),
                                hipFuncAttributeMaxDynamicSharedMemorySize, smem);
      branch_mlp<2><<<nb, 256, smem, stream>>>(A);
    }
  };

  // ---------------- SA1 ----------------
  fps_kernel<<<B, 256, 0, stream>>>(x0, x1, N, 512);
  {
    int nt = B * 512;
    ballq_kernel<<<(nt + 255) / 256, 256, 0, stream>>>(x0, x1, gi1a, B, N, 512, 32,  0.01f);
    ballq_kernel<<<(nt + 255) / 256, 256, 0, stream>>>(x0, x1, gi1b, B, N, 512, 64,  0.04f);
    ballq_kernel<<<(nt + 255) / 256, 256, 0, stream>>>(x0, x1, gi1c, B, N, 512, 128, 0.16f);
  }
  runBranch(0, 0, p0, x0, x1, gi1a, p1c, 320, 0,   512, 32,  N, 3, 6);
  runBranch(0, 3, p0, x0, x1, gi1b, p1c, 320, 64,  512, 64,  N, 3, 6);
  runBranch(0, 6, p0, x0, x1, gi1c, p1c, 320, 192, 512, 128, N, 3, 6);

  // ---------------- SA2 ----------------
  fps_kernel<<<B, 256, 0, stream>>>(x1, x2, 512, 128);
  {
    int nt = B * 128;
    ballq_kernel<<<(nt + 255) / 256, 256, 0, stream>>>(x1, x2, gi2a, B, 512, 128, 64,  0.16f);
    ballq_kernel<<<(nt + 255) / 256, 256, 0, stream>>>(x1, x2, gi2b, B, 512, 128, 128, 0.64f);
  }
  runBranch(1, 9,  p1c, x1, x2, gi2a, p2c, 512, 0,   128, 64,  512, 320, 323);
  runBranch(1, 12, p1c, x1, x2, gi2b, p2c, 512, 256, 128, 128, 512, 320, 323);

  // ---------------- SA3 ----------------
  runBranch(2, 15, p2c, x2, x2, (const int*)nullptr, (float*)d_out,
            1024, 0, 1, 128, 128, 512, 515);
}
